// LSTMEmbed_72069551227268
// MI455X (gfx1250) — matrix operand order repaired and checked by reading, not compiled, not run on hardware
//
#include <hip/hip_runtime.h>

typedef __attribute__((ext_vector_type(16))) __bf16 v16bf;
typedef __attribute__((ext_vector_type(8)))  __bf16 v8bf;
typedef __attribute__((ext_vector_type(8)))  float  v8f;

#define B_DIM 32
#define S_DIM 512
#define C_DIM 320
#define H_DIM 512
#define M_DIM (B_DIM * C_DIM)   /* 10240 */
#define K_DIM S_DIM             /* 512   */

__device__ __forceinline__ __bf16 to_bf16(float f) {
  unsigned u = __builtin_bit_cast(unsigned, f);
  unsigned r = (u + 0x7fffu + ((u >> 16) & 1u)) >> 16;   // round-to-nearest-even
  unsigned short s = (unsigned short)r;
  return __builtin_bit_cast(__bf16, s);
}

__device__ __forceinline__ float sigmoidf_(float x) {
  return 1.0f / (1.0f + __expf(-x));
}
__device__ __forceinline__ float tanhf_(float x) {
  return 2.0f * sigmoidf_(2.0f * x) - 1.0f;
}

// ---------------------------------------------------------------------------
// Kernel 1: x (B,S,C) f32  ->  At[(b*C+c)*S + s] bf16, via 32x32 LDS tile.
// Coalesced reads over c, coalesced (64B/wave) bf16 writes over s.
// grid = (C/32, S/32, B), block = 256
// ---------------------------------------------------------------------------
__global__ void convert_x_kernel(const float* __restrict__ x,
                                 __bf16* __restrict__ At) {
  __shared__ float tile[32][33];                    // +1 pad: conflict-free transpose
  const int cBase = blockIdx.x * 32;
  const int sBase = blockIdx.y * 32;
  const int b     = blockIdx.z;
  const int tr = threadIdx.x >> 5;                  // 0..7
  const int tc = threadIdx.x & 31;                  // 0..31

  const float* xp = x + ((size_t)b * S_DIM + sBase) * C_DIM + cBase;
#pragma unroll
  for (int p = 0; p < 4; ++p) {
    int sl = tr + p * 8;
    tile[sl][tc] = xp[(size_t)sl * C_DIM + tc];
  }
  __syncthreads();
#pragma unroll
  for (int p = 0; p < 4; ++p) {
    int cl = tr + p * 8;
    At[(size_t)(b * C_DIM + cBase + cl) * K_DIM + sBase + tc] = to_bf16(tile[tc][cl]);
  }
}

// ---------------------------------------------------------------------------
// Kernel 2: w_ih (4H,S) f32 -> bf16, same layout (rows already K-major).
// ---------------------------------------------------------------------------
__global__ void convert_w_kernel(const float* __restrict__ w,
                                 __bf16* __restrict__ wb) {
  const size_t i = (size_t)(blockIdx.x * blockDim.x + threadIdx.x) * 4;
  float4 v = *(const float4*)(w + i);
  unsigned short u0 = __builtin_bit_cast(unsigned short, to_bf16(v.x));
  unsigned short u1 = __builtin_bit_cast(unsigned short, to_bf16(v.y));
  unsigned short u2 = __builtin_bit_cast(unsigned short, to_bf16(v.z));
  unsigned short u3 = __builtin_bit_cast(unsigned short, to_bf16(v.w));
  ushort4 o; o.x = u0; o.y = u1; o.z = u2; o.w = u3;
  *(ushort4*)(wb + i) = o;
}

// ---------------------------------------------------------------------------
// Kernel 3: WMMA GEMM + LSTM-cell epilogue + output scramble.
// Wave tile: 32m x 32hh x 3 gates -> 12 accumulators; per k-step loads
// 2 A-frags + 6 B-frags (8KB) feeding 12 WMMAs (0.67KB/WMMA).
// Single base address per matrix; K fully unrolled so every load uses a
// constant 24-bit instruction offset (no in-loop address arithmetic).
// Block: 8 waves as 4(m) x 2(hh) -> 128m x 64hh. grid = (M/128, H/64).
// ---------------------------------------------------------------------------
__global__ void __launch_bounds__(256)
lstm_wmma_kernel(const __bf16* __restrict__ At, const __bf16* __restrict__ Wb,
                 const float* __restrict__ b_ih, const float* __restrict__ b_hh,
                 float* __restrict__ out) {
  const int wave = threadIdx.x >> 5;
  const int lane = threadIdx.x & 31;
  const int half = lane >> 4;      // which 16-lane half
  const int l15  = lane & 15;
  const int mw   = wave & 3;       // 4 m-waves
  const int hw   = wave >> 2;      // 2 hh-waves

  const int mBase  = blockIdx.x * 128 + mw * 32;
  const int hhBase = blockIdx.y * 64 + hw * 32;

  // A fragment rows (contiguous k order, same as B):
  //   elements 0..7  -> K = k + half*16 + (0..7)
  //   elements 8..15 -> K = k + half*16 + 8 + (0..7)
  const __bf16* aBase = At + (size_t)(mBase + l15) * K_DIM + half * 16;
  // B fragments: column n = gate row of w_ih; per-lane 16 consecutive K at
  // k + half*16 (contiguous k order, same as A). Gate/j streams are constant
  // offsets from this base (16KB / 1MB / 1.5MB, all within imm24).
  const __bf16* bBase = Wb + (size_t)(hhBase + l15) * K_DIM + half * 16;

  v8f acc[3][2][2] = {};   // [gate][m-frag][hh-frag]

#pragma unroll
  for (int k = 0; k < K_DIM; k += 32) {
    v16bf a[2];
#pragma unroll
    for (int i = 0; i < 2; ++i) {
      v8bf lo = *(const v8bf*)(aBase + (size_t)(i * 16) * K_DIM + k);
      v8bf hi = *(const v8bf*)(aBase + (size_t)(i * 16) * K_DIM + k + 8);
      a[i] = __builtin_shufflevector(lo, hi, 0, 1, 2, 3, 4, 5, 6, 7,
                                             8, 9, 10, 11, 12, 13, 14, 15);
    }
#pragma unroll
    for (int g = 0; g < 3; ++g) {
      const int gOff = (g == 0) ? 0 : (g == 1) ? 1024 : 1536;  // i, g, o gates
#pragma unroll
      for (int j = 0; j < 2; ++j) {
        v16bf fb = *(const v16bf*)(bBase + (size_t)(gOff + j * 16) * K_DIM + k);
#pragma unroll
        for (int i = 0; i < 2; ++i) {
          acc[g][i][j] = __builtin_amdgcn_wmma_f32_16x16x32_bf16(
              false, a[i], false, fb, (short)0, acc[g][i][j], false, false);
        }
      }
    }
  }

  // Biases per hh-fragment column (gate order i, f, g, o; f unused: c0 == 0).
  const int gateOff[3] = {0, 1024, 1536};
  float bias[3][2];
#pragma unroll
  for (int j = 0; j < 2; ++j) {
    const int hhj = hhBase + j * 16 + l15;
#pragma unroll
    for (int g = 0; g < 3; ++g)
      bias[g][j] = b_ih[gateOff[g] + hhj] + b_hh[gateOff[g] + hhj];
  }

  // C/D layout: VGPR r, lane l -> row m = base + r + 8*half, col = l15.
#pragma unroll
  for (int i = 0; i < 2; ++i) {
#pragma unroll
    for (int r = 0; r < 8; ++r) {
      const int m = mBase + i * 16 + r + 8 * half;
      const int b = m / C_DIM;
      const int c = m - b * C_DIM;
      // torch-faithful scramble: (C,B,H) flat -> reshape (B,C,H) -> transpose (B,H,C)
      const int flat = c * B_DIM + b;
      const int b2 = flat / C_DIM;
      const int c2 = flat - b2 * C_DIM;
      const size_t outBase = (size_t)b2 * (H_DIM * C_DIM) + c2;
#pragma unroll
      for (int j = 0; j < 2; ++j) {
        const int hhj = hhBase + j * 16 + l15;
        const float gi = acc[0][i][j][r] + bias[0][j];
        const float gg = acc[1][i][j][r] + bias[1][j];
        const float go = acc[2][i][j][r] + bias[2][j];
        const float cc = sigmoidf_(gi) * tanhf_(gg);
        const float hv = sigmoidf_(go) * tanhf_(cc);
        out[outBase + (size_t)hhj * C_DIM] = hv;
      }
    }
  }
}

extern "C" void kernel_launch(void* const* d_in, const int* in_sizes, int n_in,
                              void* d_out, int out_size, void* d_ws, size_t ws_size,
                              hipStream_t stream) {
  const float* x    = (const float*)d_in[0];   // (32, 512, 320)
  const float* w_ih = (const float*)d_in[1];   // (2048, 512)
  // d_in[2] = w_hh: unused (h0 == 0)
  const float* b_ih = (const float*)d_in[3];   // (2048,)
  const float* b_hh = (const float*)d_in[4];   // (2048,)
  float* out = (float*)d_out;                  // (32, 512, 320)

  __bf16* At = (__bf16*)d_ws;                                                   // 10.5 MB
  __bf16* Wb = (__bf16*)((char*)d_ws + (size_t)M_DIM * K_DIM * sizeof(__bf16)); // +2 MB

  convert_x_kernel<<<dim3(C_DIM / 32, S_DIM / 32, B_DIM), 256, 0, stream>>>(x, At);
  convert_w_kernel<<<dim3((2048 * 512) / (256 * 4)), 256, 0, stream>>>(w_ih, Wb);
  lstm_wmma_kernel<<<dim3(M_DIM / 128, H_DIM / 64), 256, 0, stream>>>(At, Wb, b_ih, b_hh, out);
}
